// WindowAttentionBlock_49641232007372
// MI455X (gfx1250) — matrix operand order repaired and checked by reading, not compiled, not run on hardware
//
#include <hip/hip_runtime.h>
#include <hip/hip_bf16.h>

typedef _Float16 half_t;
typedef half_t v16h __attribute__((ext_vector_type(16)));
typedef half_t v8h  __attribute__((ext_vector_type(8)));
typedef float  v8f  __attribute__((ext_vector_type(8)));
typedef unsigned int u32x4 __attribute__((ext_vector_type(4)));
typedef int i32x4 __attribute__((ext_vector_type(4)));
typedef int i32x8 __attribute__((ext_vector_type(8)));

#define WMMA16(a, b, c) \
  __builtin_amdgcn_wmma_f32_16x16x32_f16(false, (a), false, (b), (short)0, (c), false, false)

#if defined(__has_builtin)
#if __has_builtin(__builtin_amdgcn_tensor_load_to_lds)
#define HAVE_TDM 1
#endif
#endif

// ---------------- geometry ----------------
constexpr int CDIM  = 256;
constexpr int HIMG  = 128;
constexpr int WIMG  = 128;
constexpr int WSZ   = 8;
constexpr int LTOK  = 64;    // tokens per window (8*8)
constexpr int NHEAD = 8;
constexpr int NWIN  = 16 * 16 * 16;   // 4096 windows

// ---------------- LDS layout ----------------
constexpr int XW_ST = 65;    // floats per CHANNEL row: xw[c*65 + l]  (TDM pad = 1 dword / 64)
constexpr int XN_ST = 264;   // halfs per token row (64x256 + pad)
constexpr int VT_ST = 72;    // halfs per row (256x64 + pad), v transposed
constexpr int P_ST  = 72;    // halfs per row, per-wave prob buffer 64x64+pad
constexpr int H_ST  = 1032;  // halfs per row (64x1024 + pad) FFN hidden

constexpr int OFF_XW = 0;
constexpr int OFF_XN = OFF_XW + CDIM * XW_ST * 4;        // 66560
constexpr int OFF_Q  = OFF_XN + LTOK * XN_ST * 2;        // 100352
constexpr int OFF_K  = OFF_Q  + LTOK * XN_ST * 2;        // 134144
constexpr int OFF_VT = OFF_K  + LTOK * XN_ST * 2;        // 167936
constexpr int OFF_P  = OFF_VT + CDIM * VT_ST * 2;        // 204800
constexpr int SMEM_BYTES = OFF_P + NHEAD * LTOK * P_ST * 2;  // 278528
static_assert(SMEM_BYTES == 278528, "lds layout");
static_assert(SMEM_BYTES - OFF_K >= LTOK * H_ST * 2, "hidden alias fits");

// ---------------- fragment helpers ----------------
// A fragment (16x32 f16): lane m holds k = hi*8 + [0..7] and 16 + hi*8 + [0..7]
__device__ inline v16h load_a_frag(const half_t* __restrict__ base, int stride,
                                   int m0, int k0, int lane) {
  int m  = m0 + (lane & 15);
  int hi = lane >> 4;
  const half_t* p = base + m * stride + k0 + hi * 8;
  v8h lo = *(const v8h*)p;
  v8h hv = *(const v8h*)(p + 16);
  v16h r;
#pragma unroll
  for (int i = 0; i < 8; ++i) { r[i] = lo[i]; r[8 + i] = hv[i]; }
  return r;
}

// B fragment (32x16 f16) from a buffer laid out [n][k] row-major:
// lane n holds k = hi*8 + [0..7] and 16 + hi*8 + [0..7].
__device__ inline v16h load_b_frag(const half_t* __restrict__ base, int stride,
                                   int n0, int k0, int lane) {
  const half_t* p = base + (n0 + (lane & 15)) * stride + k0 + (lane >> 4) * 8;
  v8h lo = *(const v8h*)p;
  v8h hv = *(const v8h*)(p + 16);
  v16h r;
#pragma unroll
  for (int i = 0; i < 8; ++i) { r[i] = lo[i]; r[8 + i] = hv[i]; }
  return r;
}

// D (f32, 16x16): VGPR r -> row m0 + r + hi*8, col n0 + (lane&15)
__device__ inline void store_d_f16(half_t* __restrict__ base, int stride,
                                   int m0, int n0, int lane, const v8f& d, float scale) {
  int n  = n0 + (lane & 15);
  int r0 = m0 + (lane >> 4) * 8;
#pragma unroll
  for (int r = 0; r < 8; ++r) base[(r0 + r) * stride + n] = (half_t)(d[r] * scale);
}

// preload the 8 A fragments (K = 256) of one 16-row tile
__device__ inline void load_a_row(v16h A[8], const half_t* __restrict__ base,
                                  int stride, int m0, int lane) {
#pragma unroll
  for (int kk = 0; kk < 8; ++kk) A[kk] = load_a_frag(base, stride, m0, kk * 32, lane);
}

// LayerNorm over C=256 for 64 tokens with 256 threads (4 threads / token).
// xw is channel-major: xw[c*XW_ST + l]; dst is token-major f16.
__device__ inline void layer_norm_to_f16(const float* __restrict__ xw,
                                         half_t* __restrict__ dst,
                                         const float* __restrict__ gw,
                                         const float* __restrict__ gb, int t) {
  int l = t >> 2, qq = t & 3;
  float s = 0.f, s2 = 0.f;
#pragma unroll 8
  for (int i = 0; i < 64; ++i) {
    float v = xw[(qq * 64 + i) * XW_ST + l];
    s += v; s2 += v * v;
  }
  s  += __shfl_xor(s, 1);  s  += __shfl_xor(s, 2);
  s2 += __shfl_xor(s2, 1); s2 += __shfl_xor(s2, 2);
  float mu  = s * (1.f / 256.f);
  float var = s2 * (1.f / 256.f) - mu * mu;
  float rs  = rsqrtf(var + 1e-5f);
  half_t* drow = dst + l * XN_ST;
#pragma unroll 8
  for (int i = 0; i < 64; ++i) {
    int c = qq * 64 + i;
    drow[c] = (half_t)((xw[c * XW_ST + l] - mu) * rs * gw[c] + gb[c]);
  }
}

// ---------------- weight fp32 -> fp16 prep ----------------
__global__ void cvt_f16_kernel(const float* __restrict__ src, half_t* __restrict__ dst, int n) {
  int i = blockIdx.x * 256 + threadIdx.x;
  if (i < n) dst[i] = (half_t)src[i];
}

// ---------------- main fused window-attention block ----------------
__global__ __launch_bounds__(256)
void swin_block_kernel(const float* __restrict__ x,
                       const half_t* __restrict__ Wqkv,   // [768][256] f16
                       const float*  __restrict__ bqkv,
                       const half_t* __restrict__ Wout,   // [256][256] f16
                       const float*  __restrict__ bout,
                       const float*  __restrict__ n1w, const float* __restrict__ n1b,
                       const half_t* __restrict__ W1,     // [1024][256] f16
                       const float*  __restrict__ b1,
                       const half_t* __restrict__ W2,     // [256][1024] f16
                       const float*  __restrict__ b2,
                       const float*  __restrict__ n2w, const float* __restrict__ n2b,
                       float* __restrict__ out) {
  __shared__ __align__(16) unsigned char smem[SMEM_BYTES];
  float*  xw = (float*)(smem + OFF_XW);
  half_t* xn = (half_t*)(smem + OFF_XN);
  half_t* qb = (half_t*)(smem + OFF_Q);
  half_t* kb = (half_t*)(smem + OFF_K);
  half_t* vt = (half_t*)(smem + OFF_VT);
  half_t* pb = (half_t*)(smem + OFF_P);
  half_t* ob = qb;   // attention output reuses q region (per-head column-disjoint)
  half_t* hb = kb;   // FFN hidden reuses k/vt/p regions

  const int t    = threadIdx.x;
  const int lane = t & 31;
  const int nsel = lane & 15;
  const int hi   = lane >> 4;
  const int wv   = __builtin_amdgcn_readfirstlane(t >> 5);   // uniform wave id (SGPR)

  const int win = blockIdx.x;
  const int b   = win >> 8;
  const int ih  = (win >> 4) & 15;
  const int iw  = win & 15;
  const long base = (((long)b * CDIM) * HIMG + (long)ih * WSZ) * WIMG + iw * WSZ;

  // ---- P0: gather window into LDS (channel-major f32, stride 65) ----
#ifdef HAVE_TDM
  if (wv == 0) {
    unsigned long long ga = (unsigned long long)(const void*)(x + base);
    unsigned int ldsoff = (unsigned int)(unsigned long long)(const void*)xw;  // LDS aperture low bits
    u32x4 g0;
    g0[0] = 1u;                                   // count = 1 valid descriptor
    g0[1] = ldsoff;                               // lds_addr
    g0[2] = (unsigned int)(ga & 0xffffffffull);   // global_addr[31:0]
    g0[3] = (unsigned int)((ga >> 32) & 0x1ffffffull) | (2u << 30);  // [56:32] | type=2
    i32x8 g1;
    g1[0] = (2 << 16) | (1 << 20) | (5 << 22);    // data_size=4B, pad_enable, interval=64dw, amount=1dw
    g1[1] = (128 << 16);                          // tensor_dim0 = 128 (low16 in [31:16])
    g1[2] = (128 << 16);                          // dim0 hi16=0 ; tensor_dim1 = 128 (low16)
    g1[3] = (8 << 16);                            // dim1 hi16=0 ; tile_dim0 = 8
    g1[4] = 8 | (256 << 16);                      // tile_dim1 = 8, tile_dim2 = 256
    g1[5] = 128;                                  // tensor_dim0_stride lo32 = 128
    g1[6] = (16384 << 16);                        // d0s hi16=0 ; tensor_dim1_stride lo16 = 16384
    g1[7] = 0;                                    // tensor_dim1_stride hi32
    i32x4 g2; g2[0] = 256; g2[1] = 0; g2[2] = 0; g2[3] = 0;   // tensor_dim2 = 256
    i32x4 g3; g3[0] = 0;  g3[1] = 0; g3[2] = 0; g3[3] = 0;
#if __has_include(<hip/amd_detail/amd_gfx1250_TDM.h>)
    i32x8 g4; g4[0]=0; g4[1]=0; g4[2]=0; g4[3]=0; g4[4]=0; g4[5]=0; g4[6]=0; g4[7]=0;
    __builtin_amdgcn_tensor_load_to_lds(g0, g1, g2, g3, g4, 0);
#else
    __builtin_amdgcn_tensor_load_to_lds(g0, g1, g2, g3, 0);
#endif
    __builtin_amdgcn_s_wait_tensorcnt(0);
  }
#else
  {
    int l = t & 63;
    int r = l >> 3, cc = l & 7;
    int c0 = t >> 6;
    const float* src = x + base + (long)r * WIMG + cc;
#pragma unroll 8
    for (int i = 0; i < 64; ++i) {
      int c = c0 + i * 4;
      xw[c * XW_ST + l] = src[(long)c * (HIMG * WIMG)];
    }
  }
#endif
  __syncthreads();

  // ---- P1: LayerNorm1 -> xn (f16) ----
  layer_norm_to_f16(xw, xn, n1w, n1b, t);
  __syncthreads();

  // ---- P2: QKV = LN(x) @ Wqkv^T + b ; q scaled by 1/sqrt(hd); v transposed ----
  const float inv_sqrt_hd = 0.1767766953f;
#pragma unroll
  for (int mt = 0; mt < 4; ++mt) {
    v16h A[8];
    load_a_row(A, xn, XN_ST, mt * 16, lane);
    // q tiles
    for (int nt = wv; nt < 16; nt += 8) {
      int n0 = nt * 16;
      float bias = bqkv[n0 + nsel];
      v8f acc;
#pragma unroll
      for (int r = 0; r < 8; ++r) acc[r] = bias;
#pragma unroll
      for (int kk = 0; kk < 8; ++kk)
        acc = WMMA16(A[kk], load_b_frag(Wqkv, CDIM, n0, kk * 32, lane), acc);
      store_d_f16(qb, XN_ST, mt * 16, n0, lane, acc, inv_sqrt_hd);
    }
    // k tiles
    for (int nt = 16 + wv; nt < 32; nt += 8) {
      int n0 = nt * 16;
      float bias = bqkv[n0 + nsel];
      v8f acc;
#pragma unroll
      for (int r = 0; r < 8; ++r) acc[r] = bias;
#pragma unroll
      for (int kk = 0; kk < 8; ++kk)
        acc = WMMA16(A[kk], load_b_frag(Wqkv, CDIM, n0, kk * 32, lane), acc);
      store_d_f16(kb, XN_ST, mt * 16, n0 - 256, lane, acc, 1.f);
    }
    // v tiles (stored transposed: vt[channel][token])
    for (int nt = 32 + wv; nt < 48; nt += 8) {
      int n0 = nt * 16;
      float bias = bqkv[n0 + nsel];
      v8f acc;
#pragma unroll
      for (int r = 0; r < 8; ++r) acc[r] = bias;
#pragma unroll
      for (int kk = 0; kk < 8; ++kk)
        acc = WMMA16(A[kk], load_b_frag(Wqkv, CDIM, n0, kk * 32, lane), acc);
      int c  = (n0 - 512) + nsel;
      int r0 = mt * 16 + hi * 8;
#pragma unroll
      for (int r = 0; r < 8; ++r) vt[c * VT_ST + r0 + r] = (half_t)acc[r];
    }
  }
  __syncthreads();

  // ---- P3: per-head attention scores + register softmax (wave == head) ----
  const int head = wv;
  half_t* pw = pb + head * (LTOK * P_ST);
  {
    v8f S[4][4];
#pragma unroll
    for (int mt = 0; mt < 4; ++mt) {
      v16h a = load_a_frag(qb, XN_ST, mt * 16, head * 32, lane);
#pragma unroll
      for (int nt = 0; nt < 4; ++nt) {
        v8f z;
#pragma unroll
        for (int r = 0; r < 8; ++r) z[r] = 0.f;
        v16h bf = load_b_frag(kb, XN_ST, nt * 16, head * 32, lane);
        S[mt][nt] = WMMA16(a, bf, z);
      }
    }
#pragma unroll
    for (int mt = 0; mt < 4; ++mt) {
#pragma unroll
      for (int r = 0; r < 8; ++r) {
        float mx = fmaxf(fmaxf(S[mt][0][r], S[mt][1][r]), fmaxf(S[mt][2][r], S[mt][3][r]));
        mx = fmaxf(mx, __shfl_xor(mx, 1));
        mx = fmaxf(mx, __shfl_xor(mx, 2));
        mx = fmaxf(mx, __shfl_xor(mx, 4));
        mx = fmaxf(mx, __shfl_xor(mx, 8));
        float sum = 0.f;
#pragma unroll
        for (int nt = 0; nt < 4; ++nt) {
          float e = __expf(S[mt][nt][r] - mx);
          S[mt][nt][r] = e;
          sum += e;
        }
        sum += __shfl_xor(sum, 1);
        sum += __shfl_xor(sum, 2);
        sum += __shfl_xor(sum, 4);
        sum += __shfl_xor(sum, 8);
        float inv = 1.f / sum;
#pragma unroll
        for (int nt = 0; nt < 4; ++nt) S[mt][nt][r] *= inv;
      }
    }
#pragma unroll
    for (int mt = 0; mt < 4; ++mt)
#pragma unroll
      for (int nt = 0; nt < 4; ++nt)
#pragma unroll
        for (int r = 0; r < 8; ++r)
          pw[(mt * 16 + hi * 8 + r) * P_ST + nt * 16 + nsel] = (half_t)S[mt][nt][r];
  }

  // ---- P4: o_head = P @ V_head -> ob (only this head's columns) ----
#pragma unroll
  for (int mt = 0; mt < 4; ++mt) {
#pragma unroll
    for (int dt = 0; dt < 2; ++dt) {
      v8f acc;
#pragma unroll
      for (int r = 0; r < 8; ++r) acc[r] = 0.f;
#pragma unroll
      for (int kk = 0; kk < 2; ++kk) {
        v16h a  = load_a_frag(pw, P_ST, mt * 16, kk * 32, lane);
        v16h bf = load_b_frag(vt, VT_ST, head * 32 + dt * 16, kk * 32, lane);
        acc = WMMA16(a, bf, acc);
      }
      store_d_f16(ob, XN_ST, mt * 16, head * 32 + dt * 16, lane, acc, 1.f);
    }
  }
  __syncthreads();

  // ---- P5: out-proj + bias + residual into xw ----
#pragma unroll
  for (int mt = 0; mt < 4; ++mt) {
    v16h A[8];
    load_a_row(A, ob, XN_ST, mt * 16, lane);
    for (int nt = wv; nt < 16; nt += 8) {
      int n0 = nt * 16;
      float bias = bout[n0 + nsel];
      v8f acc;
#pragma unroll
      for (int r = 0; r < 8; ++r) acc[r] = bias;
#pragma unroll
      for (int kk = 0; kk < 8; ++kk)
        acc = WMMA16(A[kk], load_b_frag(Wout, CDIM, n0, kk * 32, lane), acc);
      int n = n0 + nsel, r0 = mt * 16 + hi * 8;
#pragma unroll
      for (int r = 0; r < 8; ++r) xw[n * XW_ST + (r0 + r)] += acc[r];
    }
  }
  __syncthreads();

  // ---- P6: LayerNorm2 -> xn ----
  layer_norm_to_f16(xw, xn, n2w, n2b, t);
  __syncthreads();

  // ---- P7: h = gelu(LN2 @ W1^T + b1) -> hb (f16) ----
#pragma unroll
  for (int mt = 0; mt < 4; ++mt) {
    v16h A[8];
    load_a_row(A, xn, XN_ST, mt * 16, lane);
    for (int nt = wv; nt < 64; nt += 8) {
      int n0 = nt * 16;
      float bias = b1[n0 + nsel];
      v8f acc;
#pragma unroll
      for (int r = 0; r < 8; ++r) acc[r] = bias;
#pragma unroll
      for (int kk = 0; kk < 8; ++kk)
        acc = WMMA16(A[kk], load_b_frag(W1, CDIM, n0, kk * 32, lane), acc);
      int n = n0 + nsel, r0 = mt * 16 + hi * 8;
#pragma unroll
      for (int r = 0; r < 8; ++r) {
        float v = acc[r];
        float g = 0.5f * v * (1.f + erff(v * 0.70710678f));   // exact gelu
        hb[(r0 + r) * H_ST + n] = (half_t)g;
      }
    }
  }
  __syncthreads();

  // ---- P8: x += h @ W2^T + b2 (K = 1024, blocked by 8 k-tiles) ----
#pragma unroll
  for (int mt = 0; mt < 4; ++mt) {
    v8f acc[2];
#pragma unroll
    for (int j = 0; j < 2; ++j) {
      float bias = b2[(wv + j * 8) * 16 + nsel];
#pragma unroll
      for (int r = 0; r < 8; ++r) acc[j][r] = bias;
    }
#pragma unroll
    for (int kb2 = 0; kb2 < 4; ++kb2) {
      v16h A[8];
#pragma unroll
      for (int kk = 0; kk < 8; ++kk)
        A[kk] = load_a_frag(hb, H_ST, mt * 16, (kb2 * 8 + kk) * 32, lane);
#pragma unroll
      for (int j = 0; j < 2; ++j) {
        int n0 = (wv + j * 8) * 16;
#pragma unroll
        for (int kk = 0; kk < 8; ++kk)
          acc[j] = WMMA16(A[kk], load_b_frag(W2, 1024, n0, (kb2 * 8 + kk) * 32, lane), acc[j]);
      }
    }
#pragma unroll
    for (int j = 0; j < 2; ++j) {
      int n = (wv + j * 8) * 16 + nsel, r0 = mt * 16 + hi * 8;
#pragma unroll
      for (int r = 0; r < 8; ++r) xw[n * XW_ST + (r0 + r)] += acc[j][r];
    }
  }
  __syncthreads();

  // ---- P9: scatter window back to output (coalesced 8-wide rows) ----
  {
    int l = t & 63;
    int r = l >> 3, cc = l & 7;
    int c0 = t >> 6;
    float* dst = out + base + (long)r * WIMG + cc;
#pragma unroll 8
    for (int i = 0; i < 64; ++i) {
      int c = c0 + i * 4;
      dst[(long)c * (HIMG * WIMG)] = xw[c * XW_ST + l];
    }
  }
}

// ---------------- launch ----------------
extern "C" void kernel_launch(void* const* d_in, const int* in_sizes, int n_in,
                              void* d_out, int out_size, void* d_ws, size_t ws_size,
                              hipStream_t stream) {
  const float* x          = (const float*)d_in[0];
  const float* in_proj_w  = (const float*)d_in[1];   // (768,256)
  const float* in_proj_b  = (const float*)d_in[2];
  const float* out_proj_w = (const float*)d_in[3];   // (256,256)
  const float* out_proj_b = (const float*)d_in[4];
  const float* norm1_w    = (const float*)d_in[5];
  const float* norm1_b    = (const float*)d_in[6];
  const float* w1         = (const float*)d_in[7];   // (1024,256)
  const float* b1         = (const float*)d_in[8];
  const float* w2         = (const float*)d_in[9];   // (256,1024)
  const float* b2         = (const float*)d_in[10];
  const float* norm2_w    = (const float*)d_in[11];
  const float* norm2_b    = (const float*)d_in[12];
  float* out = (float*)d_out;

  half_t* ws16   = (half_t*)d_ws;
  half_t* Wqkv16 = ws16;                 // 768*256
  half_t* Wout16 = Wqkv16 + 768 * 256;   // 256*256
  half_t* W1_16  = Wout16 + 256 * 256;   // 1024*256
  half_t* W2_16  = W1_16  + 1024 * 256;  // 256*1024

  cvt_f16_kernel<<<(768 * 256) / 256, 256, 0, stream>>>(in_proj_w, Wqkv16, 768 * 256);
  cvt_f16_kernel<<<(256 * 256) / 256, 256, 0, stream>>>(out_proj_w, Wout16, 256 * 256);
  cvt_f16_kernel<<<(1024 * 256) / 256, 256, 0, stream>>>(w1, W1_16, 1024 * 256);
  cvt_f16_kernel<<<(256 * 1024) / 256, 256, 0, stream>>>(w2, W2_16, 256 * 1024);

  swin_block_kernel<<<NWIN, 256, 0, stream>>>(
      x, Wqkv16, in_proj_b, Wout16, out_proj_b, norm1_w, norm1_b,
      W1_16, b1, W2_16, b2, norm2_w, norm2_b, out);
}
